// RNN_MultiRegional_29085518528926
// MI455X (gfx1250) — compile-verified, matrix operand order checked by reading, not run on hardware
//
#include <hip/hip_runtime.h>
#include <hip/hip_bf16.h>
#include <math.h>

// ---------------- problem constants ----------------
#define HID   512
#define H6    3072          // 6*HID
#define NB    16            // batch
#define TT    1000          // timesteps
#define DTC   0.01f
#define NWG   24            // persistent workgroups (24*8 waves = 192 = #tiles)

typedef __bf16 bf16_t;
typedef bf16_t v16bf __attribute__((ext_vector_type(16)));
typedef float  v8f   __attribute__((ext_vector_type(8)));

// ---------------- workspace layout (bytes) ----------------
#define WS_WT    0                 // 10*512*512 bf16 = 5,242,880 B
#define WS_RIW   5242880           // relu(inp_weight): 4*3072 f32 = 49,152 B
#define WS_HBUFB 5292032           // bf16 state double buffer: 2*16*3072*2B = 196,608 B
#define WS_WHEAD 5488640           // 512 f32
#define WS_BAR   5490688           // grid barrier

struct GridBar { unsigned count; unsigned gen; };

// region r (rows of W): list of source blocks.  blocks:
// 0:str<-str 1:str<-thal 2:str<-alm 3:gpe<-str 4:stn<-gpe
// 5:snr<-str 6:snr<-stn 7:thal<-snr 8:alm<-thal 9:alm<-alm
__constant__ int cREG_NBLK[6]   = {3,1,1,2,1,2};
__constant__ int cREG_BLK[6][3] = {{0,1,2},{3,-1,-1},{4,-1,-1},{5,6,-1},{7,-1,-1},{8,9,-1}};
__constant__ int cBLK_SRC[10]   = {0,4,5,0,1,0,2,3,4,5};

__device__ __forceinline__ unsigned short f2bf(float f) {
  unsigned u = __float_as_uint(f);
  unsigned r = u + 0x7FFFu + ((u >> 16) & 1u);   // round-to-nearest-even
  return (unsigned short)(r >> 16);
}
__device__ __forceinline__ float reluf(float x) { return x > 0.f ? x : 0.f; }

// ---------------- init: h0(bf16) = hn, reset barrier ----------------
__global__ void init_kernel(const float* __restrict__ hn, unsigned short* __restrict__ hbufb,
                            GridBar* bar) {
  int i = blockIdx.x * blockDim.x + threadIdx.x;
  if (i < NB * H6) hbufb[i] = f2bf(hn[i]);
  if (i == 0) { bar->count = 0u; bar->gen = 0u; }
}

// ---------------- pack effective weights into B-fragment layout ----------------
// layout: [block 10][ntile 32][kchunk 16][lane 32][elem 16] bf16
// lane l: n = l%16, half = l/16; elem e: e<8 -> k = half*8 + e
//                                        e>=8 -> k = 16 + half*8 + (e-8)
__global__ void pack_weights_kernel(
    const float* __restrict__ s2s_fixed,  const float* __restrict__ thal2str_w,
    const float* __restrict__ alm2str_w,  const float* __restrict__ str2gpe_w,
    const float* __restrict__ gpe2stn_w,  const float* __restrict__ str2snr_w,
    const float* __restrict__ stn2snr_w,  const float* __restrict__ snr2thal_w,
    const float* __restrict__ thal2alm_w, const float* __restrict__ alm2alm_w,
    unsigned short* __restrict__ wt) {
  int idx = blockIdx.x * blockDim.x + threadIdx.x;      // < 10*262144
  if (idx >= 10 * 262144) return;
  int bi   = idx / 262144;
  int rem  = idx - bi * 262144;
  int ntl  = rem >> 13;            // /8192
  int rem2 = rem & 8191;
  int kc   = rem2 >> 9;            // /512
  int rem3 = rem2 & 511;
  int lane = rem3 >> 4;
  int e    = rem3 & 15;
  int n_in = lane & 15, lhalf = lane >> 4;
  int k_in = (e < 8) ? (lhalf * 8 + e) : (16 + lhalf * 8 + (e - 8));
  int n = ntl * 16 + n_in;         // dst row within block (0..511)
  int k = kc * 32 + k_in;          // src col within block (0..511)
  int nk = n * HID + k;
  float val;
  switch (bi) {
    case 0: val = -s2s_fixed[nk]; break;                                      // D=-I, no relu
    case 1: val = (n >= 128 && n < 384) ? reluf(thal2str_w[nk]) : 0.f; break; // row mask
    case 2: val = (k < 359) ? reluf(alm2str_w[nk]) : 0.f; break;              // col mask (hid-153)
    case 3: val = (k >= 256) ? -reluf(str2gpe_w[nk]) : 0.f; break;            // cols [256,512)
    case 4: val = -reluf(gpe2stn_w[nk]); break;
    case 5: val = (k < 256) ? -reluf(str2snr_w[nk]) : 0.f; break;             // cols [0,256)
    case 6: val = reluf(stn2snr_w[nk]); break;
    case 7: val = -reluf(snr2thal_w[nk]); break;
    default: val = (bi == 8) ? reluf(thal2alm_w[nk])
                             : reluf(alm2alm_w[nk]) * ((k < 359) ? 1.f : -1.f); break;
  }
  wt[idx] = f2bf(val);
}

__global__ void pack_riw_kernel(const float* __restrict__ iw, float* __restrict__ riw) {
  int i = blockIdx.x * blockDim.x + threadIdx.x;
  if (i < 4 * H6) riw[i] = reluf(iw[i]);
}

// ---------------- head weight fold: whead = relu(fc1)[2560:,:] @ relu(fc2) ----------------
__global__ void head_pre_kernel(const float* __restrict__ fc1, const float* __restrict__ fc2,
                                float* __restrict__ whead) {
  int h = blockIdx.x * blockDim.x + threadIdx.x;
  if (h >= HID) return;
  float s = 0.f;
  const float* row = fc1 + (size_t)(5 * HID + h) * HID;
  for (int m = 0; m < HID; ++m) s += reluf(row[m]) * reluf(fc2[m]);
  whead[h] = s;
}

// ---------------- zero-fill x_last + x_out region ----------------
__global__ void fill_zero_kernel(float4* __restrict__ p, long n4) {
  long i = blockIdx.x * (long)blockDim.x + threadIdx.x;
  long stride = (long)gridDim.x * blockDim.x;
  float4 z = make_float4(0.f, 0.f, 0.f, 0.f);
  for (; i < n4; i += stride) p[i] = z;
}

// ---------------- persistent recurrence: 192 waves, one 16-col tile each ----------------
__global__ __launch_bounds__(256) void rnn_persistent_kernel(
    const float* __restrict__ inp, const float* __restrict__ inhib,
    const unsigned short* __restrict__ wt, const float* __restrict__ riw,
    const float* __restrict__ hn, unsigned short* __restrict__ hbufb,
    float* __restrict__ rnn_out, float* __restrict__ hn_last,
    GridBar* __restrict__ bar) {
  const int wave  = blockIdx.x * 8 + (threadIdx.x >> 5);  // 0..191
  const int lane  = threadIdx.x & 31;
  const int reg   = wave >> 5;                            // output region 0..5
  const int ntl   = wave & 31;                            // tile within region
  const int n     = wave * 16 + (lane & 15);              // global output column
  const int lhalf = lane >> 4;
  const int mrow  = lane & 15;                            // A-matrix batch row
  const float tonic_n = ((n >= 512 && n < 1536) || (n >= 2048 && n < 2560)) ? 1.f : 0.f;
  const bool inp_on = (n >= 128 && n < 384);              // strthal mask
  const int nblk = cREG_NBLK[reg];
  float rw0 = 0.f, rw1 = 0.f, rw2 = 0.f, rw3 = 0.f;
  if (inp_on) { rw0 = riw[n]; rw1 = riw[H6 + n]; rw2 = riw[2 * H6 + n]; rw3 = riw[3 * H6 + n]; }

  // this wave's fp32 state tile (C layout: m = r + 8*lhalf, col n), plus
  // the step-invariant part of the drive
  float hreg[8], dconst[8];
#pragma unroll
  for (int r = 0; r < 8; ++r) {
    const int m = r + lhalf * 8;
    hreg[r]   = hn[m * H6 + n];
    dconst[r] = inhib[m * H6 + n] + tonic_n;
  }

  for (int t = 0; t < TT; ++t) {
    const unsigned short* hcb = hbufb + ((t & 1) ? NB * H6 : 0);
    unsigned short*       hnb = hbufb + ((t & 1) ? 0 : NB * H6);

    v8f acc;
#pragma unroll
    for (int r = 0; r < 8; ++r) acc[r] = 0.f;

    for (int s = 0; s < nblk; ++s) {
      const int bi  = cREG_BLK[reg][s];
      const int src = cBLK_SRC[bi];
      const unsigned short* bt = wt + ((size_t)bi * 262144 + (size_t)ntl * 8192 + lane * 16);
      const unsigned short* at = hcb + mrow * H6 + src * HID + lhalf * 8;
#pragma unroll 4
      for (int kc = 0; kc < 16; ++kc) {
        union { uint4 q[2]; v16bf v; } B;
        const uint4* bp = (const uint4*)(bt + kc * 512);
        B.q[0] = bp[0];
        B.q[1] = bp[1];
        if (kc + 1 < 16) __builtin_prefetch(bt + (kc + 1) * 512, 0, 1);
        union { uint4 q[2]; v16bf v; } A;       // two 8-bf16 runs: k=half*8.., k=16+half*8..
        const unsigned short* ap = at + kc * 32;
        A.q[0] = *(const uint4*)(ap);
        A.q[1] = *(const uint4*)(ap + 16);
        acc = __builtin_amdgcn_wmma_f32_16x16x32_bf16(false, A.v, false, B.v,
                                                      (short)0, acc, false, false);
      }
    }

#pragma unroll
    for (int r = 0; r < 8; ++r) {
      const int m = r + lhalf * 8;                         // C-layout batch row
      float d = dconst[r];
      if (inp_on) {
        const float* ip = inp + ((size_t)m * TT + t) * 4;
        d += ip[0] * rw0 + ip[1] * rw1 + ip[2] * rw2 + ip[3] * rw3;
      }
      float hold = hreg[r];
      float hnew = hold + DTC * (-hold + acc[r] + d);
      hnew = hnew > 0.f ? hnew : 0.f;
      hreg[r] = hnew;                                      // fp32 state stays in registers
      hnb[m * H6 + n] = f2bf(hnew);                        // bf16 state for everyone's A loads
      rnn_out[((size_t)m * TT + t) * H6 + n] = hnew;
      if (t == TT - 1) hn_last[m * H6 + n] = hnew;
    }

    // ---- device-wide step barrier ----
    __syncthreads();
    if (threadIdx.x == 0) {
      __threadfence();
      unsigned prev = __hip_atomic_fetch_add(&bar->count, 1u, __ATOMIC_ACQ_REL,
                                             __HIP_MEMORY_SCOPE_AGENT);
      if (prev == NWG - 1) {
        __hip_atomic_store(&bar->count, 0u, __ATOMIC_RELAXED, __HIP_MEMORY_SCOPE_AGENT);
        __hip_atomic_fetch_add(&bar->gen, 1u, __ATOMIC_RELEASE, __HIP_MEMORY_SCOPE_AGENT);
      } else {
        while (__hip_atomic_load(&bar->gen, __ATOMIC_ACQUIRE, __HIP_MEMORY_SCOPE_AGENT)
               <= (unsigned)t) {
          __builtin_amdgcn_s_sleep(1);
        }
      }
    }
    __syncthreads();
  }
}

// ---------------- output head: one wave per (b,t) row ----------------
__global__ void head_kernel(const float* __restrict__ rnn_out,
                            const float* __restrict__ whead, float* __restrict__ out) {
  int row  = blockIdx.x * 8 + (threadIdx.x >> 5);          // 0..15999
  int lane = threadIdx.x & 31;
  if (row >= NB * TT) return;
  const float* base = rnn_out + (size_t)row * H6 + 5 * HID;
  float s = 0.f;
  for (int h = lane; h < HID; h += 32) s += base[h] * whead[h];
#pragma unroll
  for (int off = 16; off > 0; off >>= 1) s += __shfl_xor(s, off, 32);
  if (lane == 0) out[row] = 1.f / (1.f + __expf(-s));
}

// ---------------- launch ----------------
extern "C" void kernel_launch(void* const* d_in, const int* in_sizes, int n_in,
                              void* d_out, int out_size, void* d_ws, size_t ws_size,
                              hipStream_t stream) {
  const float* inp        = (const float*)d_in[0];
  const float* hn         = (const float*)d_in[1];
  const float* inhib      = (const float*)d_in[3];
  const float* thal2alm_w = (const float*)d_in[5];
  const float* thal2str_w = (const float*)d_in[6];
  const float* alm2alm_w  = (const float*)d_in[7];
  const float* alm2str_w  = (const float*)d_in[8];
  const float* str2snr_w  = (const float*)d_in[9];
  const float* str2gpe_w  = (const float*)d_in[10];
  const float* gpe2stn_w  = (const float*)d_in[11];
  const float* stn2snr_w  = (const float*)d_in[12];
  const float* snr2thal_w = (const float*)d_in[13];
  const float* inp_weight = (const float*)d_in[14];
  const float* fc1        = (const float*)d_in[15];
  const float* fc2        = (const float*)d_in[16];
  const float* s2s_fixed  = (const float*)d_in[17];

  char* ws = (char*)d_ws;
  unsigned short* wt    = (unsigned short*)(ws + WS_WT);
  float* riw            = (float*)(ws + WS_RIW);
  unsigned short* hbufb = (unsigned short*)(ws + WS_HBUFB);
  float* whead          = (float*)(ws + WS_WHEAD);
  GridBar* bar          = (GridBar*)(ws + WS_BAR);

  // d_out: [out 16000][hn_last 49152][rnn_out 49152000][x_last 49152][x_out 49152000]
  float* out_p   = (float*)d_out;
  float* hn_last = out_p + 16000;
  float* rnn_out = hn_last + 49152;
  float* x_zero  = rnn_out + 49152000;              // x_last + x_out contiguous
  const long x_n4 = (long)(49152 + 49152000) / 4;   // float4 count

  init_kernel<<<192, 256, 0, stream>>>(hn, hbufb, bar);
  pack_weights_kernel<<<10240, 256, 0, stream>>>(
      s2s_fixed, thal2str_w, alm2str_w, str2gpe_w, gpe2stn_w,
      str2snr_w, stn2snr_w, snr2thal_w, thal2alm_w, alm2alm_w, wt);
  pack_riw_kernel<<<48, 256, 0, stream>>>(inp_weight, riw);
  head_pre_kernel<<<2, 256, 0, stream>>>(fc1, fc2, whead);
  fill_zero_kernel<<<2048, 256, 0, stream>>>((float4*)x_zero, x_n4);
  rnn_persistent_kernel<<<NWG, 256, 0, stream>>>(inp, inhib, wt, riw, hn, hbufb,
                                                 rnn_out, hn_last, bar);
  head_kernel<<<2000, 256, 0, stream>>>(rnn_out, whead, out_p);
}
